// MultiHeadAttention_16930761081069
// MI455X (gfx1250) — matrix operand order repaired and checked by reading, not compiled, not run on hardware
//
#include <hip/hip_runtime.h>

#define D_MODEL   1024
#define NUM_HEADS 16
#define HEAD_DIM  64
#define BATCH     4
#define SEQ       2048
#define M_TOTAL   (BATCH * SEQ)   /* 8192 */
#define N_QKV     (3 * D_MODEL)   /* 3072 */

typedef __bf16 bf16;
typedef __attribute__((ext_vector_type(16))) __bf16 v16bf;
typedef __attribute__((ext_vector_type(8)))  __bf16 v8bf;
typedef __attribute__((ext_vector_type(8)))  float  v8f;

static __device__ __forceinline__ v8f wmma_bf16(v16bf a, v16bf b, v8f c) {
  return __builtin_amdgcn_wmma_f32_16x16x32_bf16(
      false, a, false, b, (short)0, c, false, false);
}

// A-fragment (16 rows x 32 K) from row-major source, leading dim `ld` (elements).
static __device__ __forceinline__ v16bf load_fragA(const bf16* __restrict__ base,
                                                   int row0, int k0, int ld, int lane) {
  const int r  = lane & 15;
  const int kg = lane >> 4;
  const bf16* p = base + (size_t)(row0 + r) * ld + k0 + kg * 8;
  v8bf lo = *reinterpret_cast<const v8bf*>(p);
  v8bf hi = *reinterpret_cast<const v8bf*>(p + 16);
  v16bf f;
#pragma unroll
  for (int i = 0; i < 8; ++i) { f[i] = lo[i]; f[i + 8] = hi[i]; }
  return f;
}

// B-fragment (32 K x 16 cols): B[k][n] = src[col0+n][k0+k], src row-major `ld`.
// Lane: col = L&15, kg = L>>4 -> K [8kg,8kg+8) and [16+8kg,24+8kg) (two 16B loads).
static __device__ __forceinline__ v16bf load_fragB(const bf16* __restrict__ base,
                                                   int col0, int k0, int ld, int lane) {
  const int c  = lane & 15;
  const int kh = lane >> 4;
  const bf16* p = base + (size_t)(col0 + c) * ld + k0 + kh * 8;
  v8bf lo = *reinterpret_cast<const v8bf*>(p);
  v8bf hi = *reinterpret_cast<const v8bf*>(p + 16);
  v16bf f;
#pragma unroll
  for (int i = 0; i < 8; ++i) { f[i] = lo[i]; f[i + 8] = hi[i]; }
  return f;
}

// ---------------------------------------------------------------------------
// Stage 1: fp32 -> bf16 cast
// ---------------------------------------------------------------------------
__global__ void __launch_bounds__(256)
cast_f32_to_bf16(const float* __restrict__ in, bf16* __restrict__ out, int n4) {
  int i = blockIdx.x * blockDim.x + threadIdx.x;
  if (i < n4) {
    float4 v = reinterpret_cast<const float4*>(in)[i];
    union { bf16 b[4]; unsigned long long u; } o;
    o.b[0] = (bf16)v.x; o.b[1] = (bf16)v.y; o.b[2] = (bf16)v.z; o.b[3] = (bf16)v.w;
    reinterpret_cast<unsigned long long*>(out)[i] = o.u;
  }
}

// ---------------------------------------------------------------------------
// Shared GEMM body: C[128x128] block, A[M,1024] x B[N,1024]^T, K=1024.
// Double-buffered LDS tiles filled by GLOBAL_LOAD_ASYNC_TO_LDS_B128 (ASYNCcnt),
// 8 waves each computing 32(M) x 64(N) = 2x4 WMMA tiles.
// LDS tile row stride = 40 bf16 (80B): conflict-free fragment reads.
// ---------------------------------------------------------------------------
#define TILE_LD 40
#define TILE_SZ (128 * TILE_LD)

static __device__ __forceinline__ unsigned lds_off(const void* p) {
  return (unsigned)(uintptr_t)p;   // low 32 bits of flat shared addr = LDS offset
}

static __device__ __forceinline__ void async_cp16(unsigned lds, const bf16* g) {
  asm volatile("global_load_async_to_lds_b128 %0, %1, off"
               :: "v"(lds), "v"(g) : "memory");
}

static __device__ __forceinline__ void gemm_body_128x128(
    const bf16* __restrict__ A, const bf16* __restrict__ B,
    int mblk, int nblk, bf16* tA0, bf16* tA1, bf16* tB0, bf16* tB1,
    v8f acc[8]) {
  const int t    = threadIdx.x;
  const int lane = t & 31;
  const int w    = t >> 5;
  const int wm   = w & 3;    // 32-row sub-tile
  const int wn   = w >> 2;   // 64-col sub-tile

  // each thread stages 2 x 16B chunks of A-tile and 2 x 16B of B-tile
  const int ch0  = t;          // chunk ids: row = ch>>2, ko = (ch&3)*8
  const int ch1  = t + 256;

  bf16* tA[2] = {tA0, tA1};
  bf16* tB[2] = {tB0, tB1};

  auto stage = [&](int buf, int k0) {
#pragma unroll
    for (int cc = 0; cc < 2; ++cc) {
      const int ch  = cc ? ch1 : ch0;
      const int row = ch >> 2;
      const int ko  = (ch & 3) * 8;
      async_cp16(lds_off(&tA[buf][row * TILE_LD + ko]),
                 A + (size_t)(mblk + row) * D_MODEL + k0 + ko);
      async_cp16(lds_off(&tB[buf][row * TILE_LD + ko]),
                 B + (size_t)(nblk + row) * D_MODEL + k0 + ko);
    }
  };

  stage(0, 0);
  asm volatile("s_wait_asynccnt 0x0" ::: "memory");
  __syncthreads();

  int buf = 0;
  for (int k0 = 0; k0 < D_MODEL; k0 += 32) {
    if (k0 + 32 < D_MODEL) stage(buf ^ 1, k0 + 32);

    const bf16* la = tA[buf];
    const bf16* lb = tB[buf];
    v16bf a0 = load_fragA(la, wm * 32,      0, TILE_LD, lane);
    v16bf a1 = load_fragA(la, wm * 32 + 16, 0, TILE_LD, lane);
    v16bf b0 = load_fragB(lb, wn * 64,      0, TILE_LD, lane);
    v16bf b1 = load_fragB(lb, wn * 64 + 16, 0, TILE_LD, lane);
    v16bf b2 = load_fragB(lb, wn * 64 + 32, 0, TILE_LD, lane);
    v16bf b3 = load_fragB(lb, wn * 64 + 48, 0, TILE_LD, lane);

    acc[0] = wmma_bf16(a0, b0, acc[0]);
    acc[1] = wmma_bf16(a0, b1, acc[1]);
    acc[2] = wmma_bf16(a0, b2, acc[2]);
    acc[3] = wmma_bf16(a0, b3, acc[3]);
    acc[4] = wmma_bf16(a1, b0, acc[4]);
    acc[5] = wmma_bf16(a1, b1, acc[5]);
    acc[6] = wmma_bf16(a1, b2, acc[6]);
    acc[7] = wmma_bf16(a1, b3, acc[7]);

    asm volatile("s_wait_asynccnt 0x0" ::: "memory");
    __syncthreads();
    buf ^= 1;
  }
}

// ---------------------------------------------------------------------------
// Stage 2: QKV GEMM + scatter to Q(1/8-scaled)/K [B,H,S,Hd] and V^T [B,H,Hd,S]
// ---------------------------------------------------------------------------
static __device__ __forceinline__ void qkv_store(v8f c, int m0, int n0,
                                                 const float* __restrict__ bias,
                                                 bf16* __restrict__ Qb,
                                                 bf16* __restrict__ Kb,
                                                 bf16* __restrict__ Vt,
                                                 int hg, int col) {
  const int n    = n0 + col;
  const float bv = bias[n];
  const int part = n >> 10;      // 0=Q 1=K 2=V
  const int nn   = n & 1023;
  const int h    = nn >> 6;
  const int hd   = nn & 63;
#pragma unroll
  for (int r = 0; r < 8; ++r) {
    const int m    = m0 + r + hg * 8;
    const int bidx = m >> 11;
    const int s    = m & 2047;
    const float v  = c[r] + bv;
    if (part == 0) {
      Qb[((size_t)(bidx * NUM_HEADS + h) * SEQ + s) * HEAD_DIM + hd] = (bf16)(v * 0.125f);
    } else if (part == 1) {
      Kb[((size_t)(bidx * NUM_HEADS + h) * SEQ + s) * HEAD_DIM + hd] = (bf16)v;
    } else {
      Vt[((size_t)(bidx * NUM_HEADS + h) * HEAD_DIM + hd) * SEQ + s] = (bf16)v;
    }
  }
}

__global__ void __launch_bounds__(256)
qkv_gemm(const bf16* __restrict__ X, const bf16* __restrict__ W,
         const float* __restrict__ bias,
         bf16* __restrict__ Qb, bf16* __restrict__ Kb, bf16* __restrict__ Vt) {
  __shared__ __align__(32) bf16 tA0[TILE_SZ], tA1[TILE_SZ], tB0[TILE_SZ], tB1[TILE_SZ];
  const int mblk = blockIdx.x * 128;
  const int nblk = blockIdx.y * 128;
  v8f acc[8] = {};
  gemm_body_128x128(X, W, mblk, nblk, tA0, tA1, tB0, tB1, acc);

  const int lane = threadIdx.x & 31;
  const int w    = threadIdx.x >> 5;
  const int hg = lane >> 4, col = lane & 15;
  const int m0 = mblk + (w & 3) * 32;
  const int n0 = nblk + (w >> 2) * 64;
#pragma unroll
  for (int tm = 0; tm < 2; ++tm)
#pragma unroll
    for (int tn = 0; tn < 4; ++tn)
      qkv_store(acc[tm * 4 + tn], m0 + tm * 16, n0 + tn * 16, bias, Qb, Kb, Vt, hg, col);
}

// ---------------------------------------------------------------------------
// Stage 4: output projection (fp32 out + bias)
// ---------------------------------------------------------------------------
__global__ void __launch_bounds__(256)
proj_gemm(const bf16* __restrict__ A, const bf16* __restrict__ W,
          const float* __restrict__ bias, float* __restrict__ out) {
  __shared__ __align__(32) bf16 tA0[TILE_SZ], tA1[TILE_SZ], tB0[TILE_SZ], tB1[TILE_SZ];
  const int mblk = blockIdx.x * 128;
  const int nblk = blockIdx.y * 128;
  v8f acc[8] = {};
  gemm_body_128x128(A, W, mblk, nblk, tA0, tA1, tB0, tB1, acc);

  const int lane = threadIdx.x & 31;
  const int w    = threadIdx.x >> 5;
  const int hg = lane >> 4, col = lane & 15;
  const int m0 = mblk + (w & 3) * 32;
  const int n0 = nblk + (w >> 2) * 64;
#pragma unroll
  for (int tm = 0; tm < 2; ++tm)
#pragma unroll
    for (int tn = 0; tn < 4; ++tn) {
      const v8f c = acc[tm * 4 + tn];
      const int n = n0 + tn * 16 + col;
      const float bv = bias[n];
#pragma unroll
      for (int r = 0; r < 8; ++r) {
        const int m = m0 + tm * 16 + r + hg * 8;
        out[(size_t)m * D_MODEL + n] = c[r] + bv;
      }
    }
}

// ---------------------------------------------------------------------------
// Stage 3: flash attention. One workgroup per (b, h, 64-row q block).
// Softmax parallelized 4 threads/row; padded LDS strides for bank spread.
// ---------------------------------------------------------------------------
#define SC_LD 68   /* fp32 score row stride */
#define PB_LD 72   /* bf16 P row stride (multiple of 8 -> 16B-aligned frags) */

__global__ void __launch_bounds__(256)
attention(const bf16* __restrict__ Q, const bf16* __restrict__ K,
          const bf16* __restrict__ Vt, bf16* __restrict__ ctx) {
  __shared__ __align__(32) float sc[64 * SC_LD];
  __shared__ __align__(32) bf16  pb[64 * PB_LD];
  __shared__ float pmax_s[256];
  __shared__ float psum_s[256];
  __shared__ float mrow_s[64];
  __shared__ float alpha_s[64];
  __shared__ float linv_s[64];

  const int t    = threadIdx.x;
  const int lane = t & 31;
  const int w    = t >> 5;
  const int mi   = w & 3;
  const int ni0  = (w >> 2) * 2;
  const int qb   = blockIdx.x * 64;
  const int h    = blockIdx.y;
  const int b    = blockIdx.z;
  const int row4 = t >> 2;     // softmax row
  const int q4   = t & 3;      // quarter within row

  const bf16* Qh = Q  + (size_t)(b * NUM_HEADS + h) * SEQ * HEAD_DIM;
  const bf16* Kh = K  + (size_t)(b * NUM_HEADS + h) * SEQ * HEAD_DIM;
  const bf16* Vh = Vt + (size_t)(b * NUM_HEADS + h) * HEAD_DIM * SEQ;

  const v16bf qf0 = load_fragA(Qh, qb + mi * 16, 0,  HEAD_DIM, lane);
  const v16bf qf1 = load_fragA(Qh, qb + mi * 16, 32, HEAD_DIM, lane);

  v8f o0 = {}, o1 = {};
  float mrun = -1e30f, lrun = 0.0f;   // live in threads with q4==0
  const int hg = lane >> 4, col = lane & 15;

  for (int kb = 0; kb < SEQ; kb += 64) {
    // ---- scores: S = Q . K^T
    v8f s0 = {}, s1 = {};
    {
      v16bf k0a = load_fragB(Kh, kb + ni0 * 16,       0, HEAD_DIM, lane);
      v16bf k1a = load_fragB(Kh, kb + (ni0 + 1) * 16, 0, HEAD_DIM, lane);
      s0 = wmma_bf16(qf0, k0a, s0);
      s1 = wmma_bf16(qf0, k1a, s1);
      v16bf k0b = load_fragB(Kh, kb + ni0 * 16,       32, HEAD_DIM, lane);
      v16bf k1b = load_fragB(Kh, kb + (ni0 + 1) * 16, 32, HEAD_DIM, lane);
      s0 = wmma_bf16(qf1, k0b, s0);
      s1 = wmma_bf16(qf1, k1b, s1);
    }
#pragma unroll
    for (int r = 0; r < 8; ++r) {
      const int row = mi * 16 + r + hg * 8;
      sc[row * SC_LD + ni0 * 16 + col]       = s0[r];
      sc[row * SC_LD + (ni0 + 1) * 16 + col] = s1[r];
    }
    __syncthreads();

    // ---- partial max (16 elems / thread)
    {
      float bm = -1e30f;
#pragma unroll
      for (int j = 0; j < 16; ++j) bm = fmaxf(bm, sc[row4 * SC_LD + q4 * 16 + j]);
      pmax_s[t] = bm;
    }
    __syncthreads();

    // ---- per-row reduce, running max + alpha
    if (q4 == 0) {
      const float bm = fmaxf(fmaxf(pmax_s[t], pmax_s[t + 1]),
                             fmaxf(pmax_s[t + 2], pmax_s[t + 3]));
      const float mnew = fmaxf(mrun, bm);
      alpha_s[row4] = __expf(mrun - mnew);
      mrow_s[row4]  = mnew;
      mrun = mnew;
    }
    __syncthreads();

    // ---- exp + partial sums, P -> LDS (bf16)
    {
      const float mnew = mrow_s[row4];
      float ls = 0.0f;
#pragma unroll
      for (int j = 0; j < 16; ++j) {
        const float e = __expf(sc[row4 * SC_LD + q4 * 16 + j] - mnew);
        pb[row4 * PB_LD + q4 * 16 + j] = (bf16)e;
        ls += e;
      }
      psum_s[t] = ls;
    }
    __syncthreads();

    if (q4 == 0)
      lrun = lrun * alpha_s[row4] + (psum_s[t] + psum_s[t + 1] + psum_s[t + 2] + psum_s[t + 3]);

    // ---- rescale running accumulators by alpha(row)
    float alv[8];
#pragma unroll
    for (int r = 0; r < 8; ++r) alv[r] = alpha_s[mi * 16 + r + hg * 8];
#pragma unroll
    for (int r = 0; r < 8; ++r) { o0[r] *= alv[r]; o1[r] *= alv[r]; }

    // ---- O += P . V
    {
      v16bf pa = load_fragA(pb, mi * 16, 0, PB_LD, lane);
      v16bf v0 = load_fragB(Vh, ni0 * 16,       kb, SEQ, lane);
      v16bf v1 = load_fragB(Vh, (ni0 + 1) * 16, kb, SEQ, lane);
      o0 = wmma_bf16(pa, v0, o0);
      o1 = wmma_bf16(pa, v1, o1);
      pa = load_fragA(pb, mi * 16, 32, PB_LD, lane);
      v0 = load_fragB(Vh, ni0 * 16,       kb + 32, SEQ, lane);
      v1 = load_fragB(Vh, (ni0 + 1) * 16, kb + 32, SEQ, lane);
      o0 = wmma_bf16(pa, v0, o0);
      o1 = wmma_bf16(pa, v1, o1);
    }
    __syncthreads();   // protect sc/pb for next iteration
  }

  if (q4 == 0) linv_s[row4] = 1.0f / lrun;
  __syncthreads();

#pragma unroll
  for (int r = 0; r < 8; ++r) {
    const int row   = mi * 16 + r + hg * 8;
    const float inv = linv_s[row];
    const int s     = qb + row;
    bf16* dst = ctx + (size_t)(b * SEQ + s) * D_MODEL + h * HEAD_DIM;
    dst[ni0 * 16 + col]       = (bf16)(o0[r] * inv);
    dst[(ni0 + 1) * 16 + col] = (bf16)(o1[r] * inv);
  }
}

// ---------------------------------------------------------------------------
extern "C" void kernel_launch(void* const* d_in, const int* in_sizes, int n_in,
                              void* d_out, int out_size, void* d_ws, size_t ws_size,
                              hipStream_t stream) {
  (void)in_sizes; (void)n_in; (void)out_size; (void)ws_size;
  const float* x     = (const float*)d_in[0];
  const float* Wqkv  = (const float*)d_in[1];
  const float* bqkv  = (const float*)d_in[2];
  const float* Wproj = (const float*)d_in[3];
  const float* bproj = (const float*)d_in[4];
  float* out = (float*)d_out;

  char* ws = (char*)d_ws;
  size_t off = 0;
  auto carve = [&](size_t bytes) -> char* {
    char* p = ws + off;
    off += (bytes + 255) & ~(size_t)255;
    return p;
  };
  bf16* wqkv_b  = (bf16*)carve((size_t)N_QKV * D_MODEL * 2);
  bf16* wproj_b = (bf16*)carve((size_t)D_MODEL * D_MODEL * 2);
  bf16* xb      = (bf16*)carve((size_t)M_TOTAL * D_MODEL * 2);  // reused as ctx
  bf16* Qb      = (bf16*)carve((size_t)M_TOTAL * D_MODEL * 2);
  bf16* Kb      = (bf16*)carve((size_t)M_TOTAL * D_MODEL * 2);
  bf16* Vt      = (bf16*)carve((size_t)M_TOTAL * D_MODEL * 2);
  bf16* ctx     = xb;

  int n4 = (M_TOTAL * D_MODEL) / 4;
  cast_f32_to_bf16<<<(n4 + 255) / 256, 256, 0, stream>>>(x, xb, n4);
  n4 = (N_QKV * D_MODEL) / 4;
  cast_f32_to_bf16<<<(n4 + 255) / 256, 256, 0, stream>>>(Wqkv, wqkv_b, n4);
  n4 = (D_MODEL * D_MODEL) / 4;
  cast_f32_to_bf16<<<(n4 + 255) / 256, 256, 0, stream>>>(Wproj, wproj_b, n4);

  qkv_gemm<<<dim3(M_TOTAL / 128, N_QKV / 128), 256, 0, stream>>>(
      xb, wqkv_b, bqkv, Qb, Kb, Vt);
  attention<<<dim3(SEQ / 64, NUM_HEADS, BATCH), 256, 0, stream>>>(Qb, Kb, Vt, ctx);
  proj_gemm<<<dim3(M_TOTAL / 128, D_MODEL / 128), 256, 0, stream>>>(
      ctx, wproj_b, bproj, out);
}
